// MultiHeadAttention_13065290514893
// MI455X (gfx1250) — matrix operand order repaired and checked by reading, not compiled, not run on hardware
//
#include <hip/hip_runtime.h>

typedef _Float16 v16h __attribute__((ext_vector_type(16)));
typedef float    v8f  __attribute__((ext_vector_type(8)));

#define EMBED 192
#define NH    8
#define HD    24
#define DPAD  32
#define SEQ   1024
#define BATCH 32
#define QKV_N 576
#define LDSP  40   // padded row stride (f16) for P staging

// 1/sqrt(24) * log2(e): fold softmax scale into Q and work in log2 domain.
#define QSCALE 0.294486225f

__device__ __forceinline__ v8f wmma16(v16h a, v16h b, v8f c) {
    return __builtin_amdgcn_wmma_f32_16x16x32_f16(false, a, false, b, (short)0, c, false, false);
}

// ---------------------------------------------------------------------------
// 0) Zero Q/K scratch (pads d=24..31 must be zero for valid scores).
// ---------------------------------------------------------------------------
__global__ void zero_ws_kernel(uint4* __restrict__ p, unsigned n16) {
    unsigned i = blockIdx.x * blockDim.x + threadIdx.x;
    if (i < n16) p[i] = make_uint4(0u, 0u, 0u, 0u);
}

// ---------------------------------------------------------------------------
// 1) QKV projection: qkv = x @ W_qkv + b, scatter to padded f16 Q/K/V^T.
//    One wave -> one 16x16 tile of the [32768 x 576] result, 6 k-steps
//    (fully unrolled -> 6 back-to-back v_wmma with pipelined loads).
// ---------------------------------------------------------------------------
__global__ __launch_bounds__(256) void qkv_proj_kernel(
    const float* __restrict__ x, const float* __restrict__ Wqkv,
    const float* __restrict__ bqkv,
    _Float16* __restrict__ Q, _Float16* __restrict__ K, _Float16* __restrict__ Vt)
{
    const int wave = threadIdx.x >> 5;
    const int lane = threadIdx.x & 31;
    const int half = lane >> 4;
    const int l16  = lane & 15;
    const int row0 = (blockIdx.y * 8 + wave) * 16;
    const int col  = blockIdx.x * 16 + l16;

    v8f acc = {};
#pragma unroll
    for (int k0 = 0; k0 < EMBED; k0 += 32) {
        v16h a;
        const float* xp = x + (row0 + l16) * EMBED + k0 + half * 16;
#pragma unroll
        for (int j = 0; j < 8; ++j) { a[j] = (_Float16)xp[j]; a[8 + j] = (_Float16)xp[8 + j]; }

        v16h bm;
        const float* wp = Wqkv + (k0 + half * 16) * QKV_N + col;
#pragma unroll
        for (int j = 0; j < 16; ++j) bm[j] = (_Float16)wp[j * QKV_N];

        acc = wmma16(a, bm, acc);
    }

    const float bias = bqkv[col];
    const int s   = col / EMBED;          // 0=Q, 1=K, 2=V
    const int rem = col - s * EMBED;
    const int h   = rem / HD;
    const int d   = rem - h * HD;

#pragma unroll
    for (int r = 0; r < 8; ++r) {
        const int grow = row0 + r + half * 8;
        const int bb = grow >> 10;
        const int n  = grow & (SEQ - 1);
        const float val = acc[r] + bias;
        const int bh = bb * NH + h;
        if (s == 0)      Q [(bh * SEQ + n) * DPAD + d] = (_Float16)(val * QSCALE);
        else if (s == 1) K [(bh * SEQ + n) * DPAD + d] = (_Float16)val;
        else             Vt[(bh * DPAD + d) * SEQ + n] = (_Float16)val;
    }
}

// ---------------------------------------------------------------------------
// 2) Flash attention. Block = 8 waves; each wave owns 16 queries of one (b,h).
//    Per 32-key block: 2 WMMA (scores) + online softmax + LDS transpose of P
//    + 2 WMMA (P @ V) into f32 running accumulators.
// ---------------------------------------------------------------------------
__global__ __launch_bounds__(256) void attn_kernel(
    const _Float16* __restrict__ Q, const _Float16* __restrict__ K,
    const _Float16* __restrict__ Vt, _Float16* __restrict__ O)
{
    __shared__ _Float16 plds[8 * 16 * LDSP];

    const int wave = threadIdx.x >> 5;
    const int lane = threadIdx.x & 31;
    const int half = lane >> 4;
    const int l16  = lane & 15;
    const int bh   = blockIdx.y;                       // b*8 + h
    const int q0   = blockIdx.x * 128 + wave * 16;

    const _Float16* Qb = Q  + bh * SEQ * DPAD;
    const _Float16* Kb = K  + bh * SEQ * DPAD;
    const _Float16* Vb = Vt + bh * DPAD * SEQ;
    _Float16* pl = plds + wave * 16 * LDSP;

    // Q tile in A layout (lane row = l16; K split 0-15 | 16-31)
    v16h aq;
    {
        const _Float16* qp = Qb + (q0 + l16) * DPAD + half * 16;
#pragma unroll
        for (int j = 0; j < 8; ++j) { aq[j] = qp[j]; aq[8 + j] = qp[8 + j]; }
    }

    v8f o0 = {}, o1 = {};
    float mrow[8], lrow[8];
#pragma unroll
    for (int r = 0; r < 8; ++r) { mrow[r] = -3.0e38f; lrow[r] = 0.0f; }

#pragma unroll 2
    for (int kb = 0; kb < SEQ; kb += 32) {
        // K as B operand: lane column = key, 16 contiguous head-dims per half
        v16h bk0, bk1;
        {
            const _Float16* kp0 = Kb + (kb + l16)      * DPAD + half * 16;
            const _Float16* kp1 = Kb + (kb + 16 + l16) * DPAD + half * 16;
#pragma unroll
            for (int j = 0; j < 16; ++j) { bk0[j] = kp0[j]; bk1[j] = kp1[j]; }
        }
        const v8f z = {};
        v8f s0 = wmma16(aq, bk0, z);
        v8f s1 = wmma16(aq, bk1, z);

        if (kb + 32 < SEQ) {
            __builtin_prefetch(Kb + (kb + 32 + lane) * DPAD, 0, 1);
            __builtin_prefetch(Vb + lane * SEQ + kb + 32, 0, 1);
        }

        // online softmax (row stats live per lane for its 8 rows)
#pragma unroll
        for (int r = 0; r < 8; ++r) {
            float t = fmaxf(s0[r], s1[r]);
            t = fmaxf(t, __shfl_xor(t, 1, 32));
            t = fmaxf(t, __shfl_xor(t, 2, 32));
            t = fmaxf(t, __shfl_xor(t, 4, 32));
            t = fmaxf(t, __shfl_xor(t, 8, 32));
            const float mn = fmaxf(mrow[r], t);
            const float alpha = __builtin_amdgcn_exp2f(mrow[r] - mn);
            mrow[r] = mn;
            const float p0 = __builtin_amdgcn_exp2f(s0[r] - mn);
            const float p1 = __builtin_amdgcn_exp2f(s1[r] - mn);
            s0[r] = p0; s1[r] = p1;
            float ls = p0 + p1;
            ls += __shfl_xor(ls, 1, 32);
            ls += __shfl_xor(ls, 2, 32);
            ls += __shfl_xor(ls, 4, 32);
            ls += __shfl_xor(ls, 8, 32);
            lrow[r] = lrow[r] * alpha + ls;
            o0[r] *= alpha;
            o1[r] *= alpha;
        }

        // P: C layout -> row-major f16 in per-wave LDS, read back in A layout
#pragma unroll
        for (int r = 0; r < 8; ++r) {
            const int m = r + half * 8;
            pl[m * LDSP + l16]      = (_Float16)s0[r];
            pl[m * LDSP + 16 + l16] = (_Float16)s1[r];
        }
        asm volatile("s_wait_dscnt 0" ::: "memory");
        v16h pa;
        {
            const _Float16* pp = pl + l16 * LDSP + half * 16;
#pragma unroll
            for (int j = 0; j < 8; ++j) { pa[j] = pp[j]; pa[8 + j] = pp[8 + j]; }
        }

        // V^T as B operand: lane column = head-dim, contiguous along keys
        v16h bv0, bv1;
        {
            const _Float16* vp0 = Vb + l16        * SEQ + kb + half * 16;
            const _Float16* vp1 = Vb + (16 + l16) * SEQ + kb + half * 16;
#pragma unroll
            for (int j = 0; j < 16; ++j) { bv0[j] = vp0[j]; bv1[j] = vp1[j]; }
        }
        o0 = wmma16(pa, bv0, o0);
        o1 = wmma16(pa, bv1, o1);
    }

    // normalize and store [b, n, h*24 + d] as f16 for the output projection
    const int b = bh >> 3;
    const int h = bh & 7;
#pragma unroll
    for (int r = 0; r < 8; ++r) {
        const float inv = 1.0f / lrow[r];
        const int qrow = q0 + r + half * 8;
        const int base = (b * SEQ + qrow) * EMBED + h * HD;
        O[base + l16] = (_Float16)(o0[r] * inv);
        if (l16 < 8) O[base + 16 + l16] = (_Float16)(o1[r] * inv);  // d = 16..23
    }
}

// ---------------------------------------------------------------------------
// 3) Output projection: out = O @ W_proj + b_proj (f32 result).
// ---------------------------------------------------------------------------
__global__ __launch_bounds__(256) void out_proj_kernel(
    const _Float16* __restrict__ O, const float* __restrict__ Wp,
    const float* __restrict__ bp, float* __restrict__ out)
{
    const int wave = threadIdx.x >> 5;
    const int lane = threadIdx.x & 31;
    const int half = lane >> 4;
    const int l16  = lane & 15;
    const int row0 = (blockIdx.y * 8 + wave) * 16;
    const int col  = blockIdx.x * 16 + l16;

    v8f acc = {};
#pragma unroll
    for (int k0 = 0; k0 < EMBED; k0 += 32) {
        v16h a;
        const _Float16* op = O + (row0 + l16) * EMBED + k0 + half * 16;
#pragma unroll
        for (int j = 0; j < 8; ++j) { a[j] = op[j]; a[8 + j] = op[8 + j]; }

        v16h bm;
        const float* wp = Wp + (k0 + half * 16) * EMBED + col;
#pragma unroll
        for (int j = 0; j < 16; ++j) bm[j] = (_Float16)wp[j * EMBED];

        acc = wmma16(a, bm, acc);
    }
    const float bias = bp[col];
#pragma unroll
    for (int r = 0; r < 8; ++r) {
        const int grow = row0 + r + half * 8;
        out[grow * EMBED + col] = acc[r] + bias;
    }
}

// ---------------------------------------------------------------------------
// Launch
// ---------------------------------------------------------------------------
extern "C" void kernel_launch(void* const* d_in, const int* in_sizes, int n_in,
                              void* d_out, int out_size, void* d_ws, size_t ws_size,
                              hipStream_t stream) {
    (void)in_sizes; (void)n_in; (void)out_size; (void)ws_size;

    const float* x    = (const float*)d_in[0];
    const float* Wqkv = (const float*)d_in[1];
    const float* bqkv = (const float*)d_in[2];
    const float* Wp   = (const float*)d_in[3];
    const float* bp   = (const float*)d_in[4];
    float* out = (float*)d_out;

    // Workspace layout (bytes):
    //   Q  : [32,8,1024,32] f16  -> 16 MiB
    //   K  : [32,8,1024,32] f16  -> 16 MiB
    //   Vt : [32,8,32,1024] f16  -> 16 MiB
    //   O  : [32,1024,192]  f16  -> 12 MiB
    const size_t SZ = 16777216;
    char* ws = (char*)d_ws;
    _Float16* Q  = (_Float16*)(ws);
    _Float16* K  = (_Float16*)(ws + SZ);
    _Float16* Vt = (_Float16*)(ws + 2 * SZ);
    _Float16* O  = (_Float16*)(ws + 3 * SZ);

    // Zero Q and K (their d=24..31 pads feed the score WMMAs).
    const unsigned n16 = (unsigned)((2 * SZ) / 16);
    zero_ws_kernel<<<(n16 + 255) / 256, 256, 0, stream>>>((uint4*)ws, n16);

    qkv_proj_kernel<<<dim3(QKV_N / 16, (BATCH * SEQ) / 128), 256, 0, stream>>>(
        x, Wqkv, bqkv, Q, K, Vt);

    attn_kernel<<<dim3(SEQ / 128, BATCH * NH), 256, 0, stream>>>(Q, K, Vt, O);

    out_proj_kernel<<<dim3(EMBED / 16, (BATCH * SEQ) / 128), 256, 0, stream>>>(
        O, Wp, bp, out);
}
